// HSpatialHyperGCN_13194139533747
// MI455X (gfx1250) — matrix operand order repaired and checked by reading, not compiled, not run on hardware
//
#include <hip/hip_runtime.h>
#include <hip/hip_bf16.h>

// ---------------- CDNA5 WMMA types ----------------
typedef __bf16  v16bf __attribute__((ext_vector_type(16)));
typedef __bf16  bf8v  __attribute__((ext_vector_type(8)));
typedef float   v8f   __attribute__((ext_vector_type(8)));
typedef float   v4f   __attribute__((ext_vector_type(4)));

#define NB   4
#define NCH  64
#define NN   4096     // H*W
#define NHEAD 4
#define NINTER 64
#define NOC  256      // NHEAD*NINTER
#define NOUT 64
#define NTOPK 5

// A fragment (16x32 bf16, M x K): lanes 0-15 -> K 0..7 & 16..23, lanes 16-31 -> K 8..15 & 24..31
__device__ inline v16bf load_a_frag(const __bf16* rowp, int half) {
    int koff = half ? 8 : 0;
    union { v16bf v; bf8v h[2]; } u;
    u.h[0] = *(const bf8v*)(rowp + koff);
    u.h[1] = *(const bf8v*)(rowp + koff + 16);
    return u.v;
}
// B fragment (32x16 bf16, K x N): lanes 0-15 -> K 0..7 & 16..23, lanes 16-31 -> K 8..15 & 24..31 (column K-contiguous)
__device__ inline v16bf load_b_frag(const __bf16* colp, int half) {
    int koff = half ? 8 : 0;
    union { v16bf v; bf8v h[2]; } u;
    u.h[0] = *(const bf8v*)(colp + koff);
    u.h[1] = *(const bf8v*)(colp + koff + 16);
    return u.v;
}

// ---------------- small prep kernels ----------------
__global__ __launch_bounds__(256) void k_cvt(const float* __restrict__ s, __bf16* __restrict__ d, int n) {
    int i = blockIdx.x * 256 + threadIdx.x;
    if (i < n) d[i] = (__bf16)s[i];
}

// fold BN into 64x64 conv weight: wout = w*scale(row), bout = (bconv-m)*scale + beta
__global__ __launch_bounds__(256) void k_fold(const float* __restrict__ w, const float* __restrict__ bc,
                                              const float* __restrict__ g, const float* __restrict__ be,
                                              const float* __restrict__ m, const float* __restrict__ v,
                                              __bf16* __restrict__ wout, float* __restrict__ bout) {
    int gid = blockIdx.x * 256 + threadIdx.x;       // 4096
    int o = gid >> 6, c = gid & 63;
    float scale = g[o] * rsqrtf(v[o] + 1e-5f);
    wout[gid] = (__bf16)(w[gid] * scale);
    if (c == 0) bout[o] = (bc[o] - m[o]) * scale + be[o];
}

// x (b,c,N) -> node-major bf16 xt[(b*N+n)*64+c] and channel-l2-normalized xn
__global__ __launch_bounds__(256) void k_xprep(const float* __restrict__ x,
                                               __bf16* __restrict__ xt, __bf16* __restrict__ xn) {
    int g = blockIdx.x * 256 + threadIdx.x;         // 0..B*N-1
    int b = g >> 12, n = g & (NN - 1);
    const float* xb = x + (size_t)b * NCH * NN + n;
    float s = 0.f;
    for (int c = 0; c < NCH; ++c) { float vv = xb[(size_t)c * NN]; s += vv * vv; }
    float inv = 1.0f / fmaxf(sqrtf(s), 1e-12f);
    for (int c0 = 0; c0 < NCH; c0 += 8) {
        bf8v t, nv;
#pragma unroll
        for (int j = 0; j < 8; ++j) {
            float vv = xb[(size_t)(c0 + j) * NN];
            t[j]  = (__bf16)vv;
            nv[j] = (__bf16)(vv * inv);
        }
        *(bf8v*)(xt + (size_t)g * NCH + c0) = t;
        *(bf8v*)(xn + (size_t)g * NCH + c0) = nv;
    }
}

// ---------------- generic WMMA GEMM:  D[o,n] = act( sum_k A[o,k]*B[n,k] + bias[o] ) ----------------
// A: (Mtot x K) bf16 K-contiguous.  B: per-batch (N x K) bf16 K-contiguous.
// MODE 0: f32 out, (b,head,n,f) layout  (Mtot=256)
// MODE 1: bf16 out, node-major (b,n,o)  (Mtot=64)
// MODE 2: f32 out, channel-major (b,o,n) (Mtot=64)  -> final d_out
template<int MODE, bool RELU, int K>
__global__ __launch_bounds__(256) void k_gemm(const __bf16* __restrict__ A, const __bf16* __restrict__ Bm,
                                              const float* __restrict__ bias, void* __restrict__ outp) {
    const int lane = threadIdx.x & 31;
    const int wave = threadIdx.x >> 5;
    const int half = lane >> 4;
    const int l15  = lane & 15;
    const int nt = blockIdx.x * 8 + wave;           // node tile
    const int mt = blockIdx.y;                      // output-channel tile
    const int b  = blockIdx.z;
    const __bf16* arow = A  + (size_t)(mt * 16 + l15) * K;
    const __bf16* bcol = Bm + (size_t)b * NN * K + (size_t)(nt * 16 + l15) * K;
    v8f c = {};
#pragma unroll
    for (int ks = 0; ks < K; ks += 32) {
        v16bf af = load_a_frag(arow + ks, half);
        v16bf bf_ = load_b_frag(bcol + ks, half);
        c = __builtin_amdgcn_wmma_f32_16x16x32_bf16(false, af, false, bf_, (short)0, c, false, false);
    }
    const int n = nt * 16 + l15;
    const int obase = mt * 16 + half * 8;           // 8 contiguous output channels per lane
    float r8[8];
#pragma unroll
    for (int r = 0; r < 8; ++r) {
        float val = c[r] + bias[obase + r];
        if (RELU) val = fmaxf(val, 0.f);
        r8[r] = val;
    }
    if constexpr (MODE == 0) {
        float* p = (float*)outp + (size_t)b * NOC * NN + (size_t)(obase >> 6) * NN * 64
                 + (size_t)n * 64 + (obase & 63);
        *(v4f*)(p)     = (v4f){r8[0], r8[1], r8[2], r8[3]};
        *(v4f*)(p + 4) = (v4f){r8[4], r8[5], r8[6], r8[7]};
    } else if constexpr (MODE == 1) {
        bf8v pk;
#pragma unroll
        for (int r = 0; r < 8; ++r) pk[r] = (__bf16)r8[r];
        *(bf8v*)((__bf16*)outp + ((size_t)b * NN + n) * NOUT + obase) = pk;
    } else {
        float* p = (float*)outp + ((size_t)b * NOUT + obase) * NN + n;
#pragma unroll
        for (int r = 0; r < 8; ++r) p[(size_t)r * NN] = r8[r];
    }
}

// ---------------- per-row l2 norm over 64 feats (q,k), in place ----------------
__global__ __launch_bounds__(256) void k_rownorm(float* __restrict__ t) {
    int lane = threadIdx.x & 31;
    int row = blockIdx.x * 8 + (threadIdx.x >> 5);  // rows = B*NHEAD*NN
    float* p = t + (size_t)row * 64;
    float a = p[lane], b2 = p[lane + 32];
    float s = a * a + b2 * b2;
#pragma unroll
    for (int m = 16; m; m >>= 1) s += __shfl_xor(s, m, 32);
    float inv = 1.0f / fmaxf(sqrtf(s), 1e-12f);
    p[lane] = a * inv; p[lane + 32] = b2 * inv;
}

// ---------------- fused cosine-sim (WMMA) + distributed top-5; sim never materialized ----------------
// Block = 256 threads = 8 waves. Each chunk: 8 waves compute 16x16 WMMA tiles -> 16x128 LDS strip.
// Scan: 16 threads per row, each owns 8 columns/chunk with a private top-5; exact merge at the end.
__global__ __launch_bounds__(256) void k_topk(const __bf16* __restrict__ xn, int* __restrict__ idx) {
    __shared__ float tile[16 * 128];
    __shared__ float cv[256 * NTOPK];
    __shared__ int   ci[256 * NTOPK];
    const int tid = threadIdx.x;
    const int lane = tid & 31, wave = tid >> 5, half = lane >> 4, l15 = lane & 15;
    const int row = tid >> 4, seg = tid & 15;       // scan-phase mapping
    const int b = blockIdx.x >> 8;
    const int i0 = (blockIdx.x & 255) * 16;
    const __bf16* xb = xn + (size_t)b * NN * NCH;
    const __bf16* arow = xb + (size_t)(i0 + l15) * NCH;
    float tv[NTOPK]; int ti[NTOPK];
#pragma unroll
    for (int t = 0; t < NTOPK; ++t) { tv[t] = -3.0e38f; ti[t] = 0x7fffffff; }
    for (int chunk = 0; chunk < NN / 128; ++chunk) {
        const int cbase = chunk * 128;
        const __bf16* bcol = xb + (size_t)(cbase + wave * 16 + l15) * NCH;
        if (chunk + 1 < NN / 128)                    // prefetch next chunk's B columns
            __builtin_prefetch(bcol + 128 * NCH, 0, 3);
        v8f c = {};
#pragma unroll
        for (int ks = 0; ks < NCH; ks += 32) {
            v16bf af = load_a_frag(arow + ks, half);
            v16bf bf_ = load_b_frag(bcol + ks, half);
            c = __builtin_amdgcn_wmma_f32_16x16x32_bf16(false, af, false, bf_, (short)0, c, false, false);
        }
#pragma unroll
        for (int r = 0; r < 8; ++r)
            tile[(r + half * 8) * 128 + wave * 16 + l15] = c[r];
        __syncthreads();
        // each thread scans its 8 columns of its row (ascending col order -> strict > keeps lax.top_k ties)
        const float* sp = &tile[row * 128 + seg * 8];
        v4f va = *(const v4f*)sp;
        v4f vb = *(const v4f*)(sp + 4);
        float vals[8] = {va[0], va[1], va[2], va[3], vb[0], vb[1], vb[2], vb[3]};
        const int cb = cbase + seg * 8;
#pragma unroll
        for (int j = 0; j < 8; ++j) {
            float v = vals[j];
            if (v > tv[NTOPK - 1]) {
                tv[NTOPK - 1] = v; ti[NTOPK - 1] = cb + j;
#pragma unroll
                for (int s = NTOPK - 1; s > 0; --s)
                    if (tv[s] > tv[s - 1]) {
                        float tf = tv[s]; tv[s] = tv[s - 1]; tv[s - 1] = tf;
                        int tt = ti[s]; ti[s] = ti[s - 1]; ti[s - 1] = tt;
                    }
            }
        }
        __syncthreads();
    }
    // dump per-thread candidates, then one exact 80-way merge per row
#pragma unroll
    for (int t = 0; t < NTOPK; ++t) { cv[tid * NTOPK + t] = tv[t]; ci[tid * NTOPK + t] = ti[t]; }
    __syncthreads();
    if (seg == 0) {
        float fv[NTOPK]; int fi[NTOPK];
#pragma unroll
        for (int t = 0; t < NTOPK; ++t) { fv[t] = -3.0e38f; fi[t] = 0x7fffffff; }
        for (int s = 0; s < 16; ++s) {
#pragma unroll
            for (int e = 0; e < NTOPK; ++e) {
                float v = cv[(row * 16 + s) * NTOPK + e];
                int   d = ci[(row * 16 + s) * NTOPK + e];
                if (v > fv[NTOPK - 1] || (v == fv[NTOPK - 1] && d < fi[NTOPK - 1])) {
                    fv[NTOPK - 1] = v; fi[NTOPK - 1] = d;
#pragma unroll
                    for (int t = NTOPK - 1; t > 0; --t)
                        if (fv[t] > fv[t - 1] || (fv[t] == fv[t - 1] && fi[t] < fi[t - 1])) {
                            float tf = fv[t]; fv[t] = fv[t - 1]; fv[t - 1] = tf;
                            int tt = fi[t]; fi[t] = fi[t - 1]; fi[t - 1] = tt;
                        }
                }
            }
        }
#pragma unroll
        for (int t = 0; t < NTOPK; ++t)
            idx[((size_t)b * NN + i0 + row) * NTOPK + t] = fi[t];
    }
}

// ---------------- "Laplacian": deg==K+1 uniformly -> pure gather: (self + sum_topk)/6 ----------------
__global__ __launch_bounds__(64) void k_laplacian(const float* __restrict__ kin, const float* __restrict__ vin,
                                                  const int* __restrict__ idx,
                                                  float* __restrict__ kout, float* __restrict__ vout) {
    const int i = blockIdx.x & (NN - 1);
    const int b = blockIdx.x >> 12;
    const int f = threadIdx.x;
    int nb[NTOPK];
#pragma unroll
    for (int t = 0; t < NTOPK; ++t) nb[t] = idx[((size_t)b * NN + i) * NTOPK + t];
    const float inv6 = 1.0f / 6.0f;
    for (int h = 0; h < NHEAD; ++h) {
        const size_t base = (size_t)(b * NHEAD + h) * NN * 64;
        float ks = kin[base + (size_t)i * 64 + f];
        float vs = vin[base + (size_t)i * 64 + f];
#pragma unroll
        for (int t = 0; t < NTOPK; ++t) {
            ks += kin[base + (size_t)nb[t] * 64 + f];
            vs += vin[base + (size_t)nb[t] * 64 + f];
        }
        kout[base + (size_t)i * 64 + f] = ks * inv6;
        vout[base + (size_t)i * 64 + f] = vs * inv6;
    }
}

// ---------------- kv[b,h,f] = sum_n k*v ----------------
__global__ __launch_bounds__(256) void k_kvred(const float* __restrict__ kl, const float* __restrict__ vl,
                                               float* __restrict__ kv) {
    __shared__ float sd[256];
    const int bh = blockIdx.x, tid = threadIdx.x;
    const int f = tid & 63, part = tid >> 6;
    const size_t base = (size_t)bh * NN * 64;
    float s = 0.f;
    for (int n = part; n < NN; n += 4)
        s += kl[base + (size_t)n * 64 + f] * vl[base + (size_t)n * 64 + f];
    sd[tid] = s; __syncthreads();
    if (tid < 64) kv[bh * 64 + tid] = sd[tid] + sd[tid + 64] + sd[tid + 128] + sd[tid + 192];
}

// ---------------- hy[b,n,h*64+f] = q * kv  (bf16, node-major for next GEMM) ----------------
__global__ __launch_bounds__(256) void k_hydra(const float* __restrict__ q, const float* __restrict__ kv,
                                               __bf16* __restrict__ hy) {
    int t = blockIdx.x * 256 + threadIdx.x;         // B*4*4096*64 = 2^22
    int f = t & 63, h = (t >> 6) & 3, n = (t >> 8) & (NN - 1), b = t >> 20;
    float val = q[((size_t)(b * NHEAD + h) * NN + n) * 64 + f] * kv[(b * NHEAD + h) * 64 + f];
    hy[((size_t)b * NN + n) * NOC + h * 64 + f] = (__bf16)val;
}

// ---------------- host-side orchestration ----------------
extern "C" void kernel_launch(void* const* d_in, const int* in_sizes, int n_in,
                              void* d_out, int out_size, void* d_ws, size_t ws_size,
                              hipStream_t stream) {
    (void)in_sizes; (void)n_in; (void)out_size; (void)ws_size;
    const float* x   = (const float*)d_in[0];
    const float* wk  = (const float*)d_in[1];  const float* bk = (const float*)d_in[2];
    const float* wq  = (const float*)d_in[3];  const float* bq = (const float*)d_in[4];
    const float* wv  = (const float*)d_in[5];  const float* bv = (const float*)d_in[6];
    const float* wp  = (const float*)d_in[7];  const float* bp = (const float*)d_in[8];
    const float* w1  = (const float*)d_in[9];  const float* b1 = (const float*)d_in[10];
    const float* w2  = (const float*)d_in[11]; const float* b2 = (const float*)d_in[12];
    const float* g1 = (const float*)d_in[13]; const float* be1 = (const float*)d_in[14];
    const float* m1 = (const float*)d_in[15]; const float* v1  = (const float*)d_in[16];
    const float* g2 = (const float*)d_in[17]; const float* be2 = (const float*)d_in[18];
    const float* m2 = (const float*)d_in[19]; const float* v2  = (const float*)d_in[20];

    char* w = (char*)d_ws;
    size_t off = 0;
    auto alloc = [&](size_t bytes) { size_t r = off; off = (off + bytes + 255) & ~(size_t)255; return r; };
    __bf16* xt   = (__bf16*)(w + alloc((size_t)NB * NN * NCH * 2));
    __bf16* xn   = (__bf16*)(w + alloc((size_t)NB * NN * NCH * 2));
    __bf16* wkb  = (__bf16*)(w + alloc((size_t)NOC * NCH * 2));
    __bf16* wqb  = (__bf16*)(w + alloc((size_t)NOC * NCH * 2));
    __bf16* wvb  = (__bf16*)(w + alloc((size_t)NOC * NCH * 2));
    __bf16* wpb  = (__bf16*)(w + alloc((size_t)NOUT * NOC * 2));
    __bf16* w1b  = (__bf16*)(w + alloc((size_t)NOUT * NOUT * 2));
    __bf16* w2b  = (__bf16*)(w + alloc((size_t)NOUT * NOUT * 2));
    float*  b1f  = (float*)(w + alloc(NOUT * 4));
    float*  b2f  = (float*)(w + alloc(NOUT * 4));
    float*  qf   = (float*)(w + alloc((size_t)NB * NOC * NN * 4));
    float*  kf   = (float*)(w + alloc((size_t)NB * NOC * NN * 4));
    float*  vf   = (float*)(w + alloc((size_t)NB * NOC * NN * 4));
    float*  klap = (float*)(w + alloc((size_t)NB * NOC * NN * 4));
    float*  vlap = (float*)(w + alloc((size_t)NB * NOC * NN * 4));
    int*    idx  = (int*)  (w + alloc((size_t)NB * NN * NTOPK * 4));
    float*  kvf  = (float*)(w + alloc((size_t)NB * NHEAD * 64 * 4));
    __bf16* hyb  = (__bf16*)(w + alloc((size_t)NB * NN * NOC * 2));
    __bf16* bu1  = (__bf16*)(w + alloc((size_t)NB * NN * NOUT * 2));
    __bf16* bu2  = (__bf16*)(w + alloc((size_t)NB * NN * NOUT * 2));

    // 1) weight conversion / BN folding
    k_cvt<<<64, 256, 0, stream>>>(wk, wkb, NOC * NCH);
    k_cvt<<<64, 256, 0, stream>>>(wq, wqb, NOC * NCH);
    k_cvt<<<64, 256, 0, stream>>>(wv, wvb, NOC * NCH);
    k_cvt<<<64, 256, 0, stream>>>(wp, wpb, NOUT * NOC);
    k_fold<<<16, 256, 0, stream>>>(w1, b1, g1, be1, m1, v1, w1b, b1f);
    k_fold<<<16, 256, 0, stream>>>(w2, b2, g2, be2, m2, v2, w2b, b2f);
    // 2) x -> node-major bf16 (+ channel-normalized copy for similarity)
    k_xprep<<<(NB * NN) / 256, 256, 0, stream>>>(x, xt, xn);
    // 3) q/k/v 1x1 convs (WMMA)
    {
        dim3 g(NN / 128, NOC / 16, NB);
        k_gemm<0, false, NCH><<<g, 256, 0, stream>>>(wkb, xt, bk, kf);
        k_gemm<0, false, NCH><<<g, 256, 0, stream>>>(wqb, xt, bq, qf);
        k_gemm<0, false, NCH><<<g, 256, 0, stream>>>(wvb, xt, bv, vf);
    }
    // 4) l2 norm rows of q and k
    k_rownorm<<<(NB * NHEAD * NN) / 8, 256, 0, stream>>>(qf);
    k_rownorm<<<(NB * NHEAD * NN) / 8, 256, 0, stream>>>(kf);
    // 5) fused cosine-sim + top-5 (WMMA, sim kept in 8KB LDS chunks, parallel scan)
    k_topk<<<NB * (NN / 16), 256, 0, stream>>>(xn, idx);
    // 6) hypergraph Laplacian (pure gather: deg==6 uniformly)
    k_laplacian<<<NB * NN, 64, 0, stream>>>(kf, vf, idx, klap, vlap);
    // 7) kv reduction + hydra scaling
    k_kvred<<<NB * NHEAD, 256, 0, stream>>>(klap, vlap, kvf);
    k_hydra<<<(NB * NHEAD * NN * 64) / 256, 256, 0, stream>>>(qf, kvf, hyb);
    // 8) projection + 2x (conv + folded-BN + ReLU), all WMMA
    {
        dim3 g(NN / 128, NOUT / 16, NB);
        k_gemm<1, false, NOC><<<g, 256, 0, stream>>>(wpb, hyb, bp, bu1);
        k_gemm<1, true,  NOUT><<<g, 256, 0, stream>>>(w1b, bu1, b1f, bu2);
        k_gemm<2, true,  NOUT><<<g, 256, 0, stream>>>(w2b, bu2, b2f, d_out);
    }
}
